// Lstm_24180665876729
// MI455X (gfx1250) — matrix operand order repaired and checked by reading, not compiled, not run on hardware
//
#include <hip/hip_runtime.h>

typedef __attribute__((ext_vector_type(16))) _Float16     v16h;
typedef __attribute__((ext_vector_type(8)))  float        v8f;
typedef __attribute__((ext_vector_type(8)))  unsigned int v8u;

#define SEQ    48
#define HID    16
#define NFEA   4
#define NSEQS  (32 * 2048)     // BS * NODES
#define NTILES (NSEQS / 16)    // 4096 waves, 16 sequences each

// ---- fast device math: native v_exp_f32 / v_rcp_f32 (TRANS ops) ----
__device__ __forceinline__ float fexp2_(float x) { return __builtin_amdgcn_exp2f(x); }
__device__ __forceinline__ float frcp_(float x)  { return __builtin_amdgcn_rcpf(x); }
__device__ __forceinline__ float sigm_(float x)  { return frcp_(1.0f + fexp2_(-1.44269504f * x)); }
__device__ __forceinline__ float tanh_(float x)  { return 1.0f - 2.0f * frcp_(1.0f + fexp2_(2.88539008f * x)); }

// pack two f32 -> one dword of two f16 (v_cvt_pk_rtz_f16_f32)
__device__ __forceinline__ unsigned int pkh_(float a, float b) {
  auto h = __builtin_amdgcn_cvt_pkrtz(a, b);
  return __builtin_bit_cast(unsigned int, h);
}

// ds_swizzle SWAPX16: lane L <-> lane L^16 (group-of-32 mode, xor=0x10, and=0x1f)
#define SWZ16(x) __builtin_amdgcn_ds_swizzle((int)(x), 0x401F)

__device__ __forceinline__ v8f wmma16(v16h A, v16h B, v8f C) {
  // D = A(16x32 f16) x B(32x16 f16) + C(16x16 f32)
  return __builtin_amdgcn_wmma_f32_16x16x32_f16(false, A, false, B, (short)0, C, false, false);
}

__global__ __launch_bounds__(256)
void lstm2_wmma_kernel(const float* __restrict__ feat,
                       const float* __restrict__ extra,
                       const float* __restrict__ Wih0, const float* __restrict__ Whh0,
                       const float* __restrict__ bih0, const float* __restrict__ bhh0,
                       const float* __restrict__ Wih1, const float* __restrict__ Whh1,
                       const float* __restrict__ bih1, const float* __restrict__ bhh1,
                       const float* __restrict__ Wlin, const float* __restrict__ blin,
                       float* __restrict__ out)
{
  const int lane = threadIdx.x & 31;
  const int n    = lane & 15;        // sequence column within this wave's 16-seq tile
  const int hi   = lane >> 4;        // lane half: selects K sub-block (A/B) and M rows (C/D)
  const int mb   = hi ? 8 : 0;
  const int tile = (blockIdx.x * blockDim.x + threadIdx.x) >> 5;

  // ---------------- preload weights into WMMA A-layout -------------------
  // A tile t covers gates [16t..16t+15]. K=0..15 -> input weights, K=16..31 -> recurrent.
  // Per-lane A halfs 0..15 <-> K = 16*hi+j (contiguous).
  v16h a0[4], a1[4];
  v8f  cb0[4], cb1[4];
#pragma unroll
  for (int t = 0; t < 4; ++t) {
    const int m = t * 16 + n;        // gate row
#pragma unroll
    for (int j = 0; j < 16; ++j) {
      const int k = j;
      a0[t][j] = (_Float16)(hi ? Whh0[m * HID + k] : ((k < NFEA) ? Wih0[m * NFEA + k] : 0.0f));
      a1[t][j] = (_Float16)(hi ? Whh1[m * HID + k] : Wih1[m * HID + k]);
    }
#pragma unroll
    for (int r = 0; r < 8; ++r) {    // bias in C layout: vgpr r <-> gate row mb+r
      cb0[t][r] = bih0[t * 16 + mb + r] + bhh0[t * 16 + mb + r];
      cb1[t][r] = bih1[t * 16 + mb + r] + bhh1[t * 16 + mb + r];
    }
  }

  // ---------------- per-wave LSTM state (registers only) -----------------
  // pX = own half of h packed f16 (4 dwords), qX = opposite half via SWAPX16.
  unsigned pA[4] = {0, 0, 0, 0}, qA[4] = {0, 0, 0, 0};   // layer-0 hidden h1
  unsigned pB[4] = {0, 0, 0, 0}, qB[4] = {0, 0, 0, 0};   // layer-1 hidden h2
  v8f cA = {0, 0, 0, 0, 0, 0, 0, 0};
  v8f cB = {0, 0, 0, 0, 0, 0, 0, 0};
  float acc = 0.0f;                                      // linear-head partial

  const int s = tile * 16 + n;                           // global sequence id
  const float* fx = feat  + (size_t)s * SEQ;
  const float* ex = extra + (size_t)s * SEQ * (NFEA - 1);

#pragma unroll 1
  for (int t = 0; t < SEQ; ++t) {
    // ---- layer 0: B = [x_t in K=0..3, zeros, h1_{t-1} in K=16..31] ----
    const float x0 = fx[t];
    const float x1 = ex[t * 3 + 0], x2 = ex[t * 3 + 1], x3 = ex[t * 3 + 2];
    v8u bu;
    bu[0] = hi ? qA[0] : pkh_(x0, x1);
    bu[1] = hi ? qA[1] : pkh_(x2, x3);
    bu[2] = hi ? qA[2] : 0u;
    bu[3] = hi ? qA[3] : 0u;
    bu[4] = hi ? pA[0] : 0u;
    bu[5] = hi ? pA[1] : 0u;
    bu[6] = hi ? pA[2] : 0u;
    bu[7] = hi ? pA[3] : 0u;
    v16h B0 = __builtin_bit_cast(v16h, bu);

    v8f gi = wmma16(a0[0], B0, cb0[0]);
    v8f gf = wmma16(a0[1], B0, cb0[1]);
    v8f gg = wmma16(a0[2], B0, cb0[2]);
    v8f go = wmma16(a0[3], B0, cb0[3]);

    float hA[8];
#pragma unroll
    for (int r = 0; r < 8; ++r) {
      const float iv = sigm_(gi[r]);
      const float fv = sigm_(gf[r]);
      const float gv = tanh_(gg[r]);
      const float ov = sigm_(go[r]);
      cA[r] = fv * cA[r] + iv * gv;
      hA[r] = ov * tanh_(cA[r]);
    }
    unsigned paN[4], qaN[4];
#pragma unroll
    for (int j = 0; j < 4; ++j) paN[j] = pkh_(hA[2 * j], hA[2 * j + 1]);
#pragma unroll
    for (int j = 0; j < 4; ++j) qaN[j] = (unsigned)SWZ16(paN[j]);

    // ---- layer 1: B = [h1_t in K=0..15, h2_{t-1} in K=16..31] ----
    bu[0] = hi ? qB[0] : paN[0];
    bu[1] = hi ? qB[1] : paN[1];
    bu[2] = hi ? qB[2] : paN[2];
    bu[3] = hi ? qB[3] : paN[3];
    bu[4] = hi ? pB[0] : qaN[0];
    bu[5] = hi ? pB[1] : qaN[1];
    bu[6] = hi ? pB[2] : qaN[2];
    bu[7] = hi ? pB[3] : qaN[3];
    v16h B1 = __builtin_bit_cast(v16h, bu);

    gi = wmma16(a1[0], B1, cb1[0]);
    gf = wmma16(a1[1], B1, cb1[1]);
    gg = wmma16(a1[2], B1, cb1[2]);
    go = wmma16(a1[3], B1, cb1[3]);

    float hB[8];
#pragma unroll
    for (int r = 0; r < 8; ++r) {
      const float iv = sigm_(gi[r]);
      const float fv = sigm_(gf[r]);
      const float gv = tanh_(gg[r]);
      const float ov = sigm_(go[r]);
      cB[r] = fv * cB[r] + iv * gv;
      hB[r] = ov * tanh_(cB[r]);
    }

    // ---- linear head: acc += h2_t[hid] * Wlin[t*16+hid] (this lane's 8 rows) ----
    const float* wl = Wlin + t * HID + mb;
#pragma unroll
    for (int r = 0; r < 8; ++r) acc = fmaf(hB[r], wl[r], acc);

    // ---- repack states for next step ----
#pragma unroll
    for (int j = 0; j < 4; ++j) {
      pB[j] = pkh_(hB[2 * j], hB[2 * j + 1]);
      qB[j] = (unsigned)SWZ16(pB[j]);
      pA[j] = paN[j];
      qA[j] = qaN[j];
    }
  }

  // fold the two lane halves (hid 0-7 + hid 8-15) and write one value per sequence
  const float other = __builtin_bit_cast(float, SWZ16(__builtin_bit_cast(int, acc)));
  const float tot   = acc + other;
  if (!hi) out[s] = tot + blin[0];
}

extern "C" void kernel_launch(void* const* d_in, const int* in_sizes, int n_in,
                              void* d_out, int out_size, void* d_ws, size_t ws_size,
                              hipStream_t stream)
{
  const float* feat  = (const float*)d_in[0];
  const float* extra = (const float*)d_in[1];
  const float* Wih0  = (const float*)d_in[2];
  const float* Whh0  = (const float*)d_in[3];
  const float* bih0  = (const float*)d_in[4];
  const float* bhh0  = (const float*)d_in[5];
  const float* Wih1  = (const float*)d_in[6];
  const float* Whh1  = (const float*)d_in[7];
  const float* bih1  = (const float*)d_in[8];
  const float* bhh1  = (const float*)d_in[9];
  const float* Wlin  = (const float*)d_in[10];
  const float* blin  = (const float*)d_in[11];
  float* out = (float*)d_out;

  // 4096 tiles of 16 sequences, 1 wave32 per tile, 8 waves per 256-thread block
  dim3 block(256);
  dim3 grid(NTILES / 8);
  lstm2_wmma_kernel<<<grid, block, 0, stream>>>(feat, extra, Wih0, Whh0, bih0, bhh0,
                                                Wih1, Whh1, bih1, bhh1, Wlin, blin, out);
}
